// MESM_27745488732759
// MI455X (gfx1250) — matrix operand order repaired and checked by reading, not compiled, not run on hardware
//
#include <hip/hip_runtime.h>
#include <hip/hip_bf16.h>
#include <cmath>

typedef __attribute__((ext_vector_type(16))) _Float16 v16h;
typedef __attribute__((ext_vector_type(8)))  float    v8f;

#define GT_M 128
#define GT_N 64
#define GT_K 32
#define LDSU 18   // dwords per LDS row (= 36 halves): bank-conflict-free fragment reads

__device__ inline unsigned packh(float a, float b) {
    union { _Float16 h[2]; unsigned u; } P;
    P.h[0] = (_Float16)a;
    P.h[1] = (_Float16)b;
    return P.u;
}

// ---------------------------------------------------------------------------
// Generic fused GEMM:  C[M,N] = act( scale * A[M,K] (x) B  + bias + resid )
//   transB==0 : B is [K,N] row-major (ldb = row stride)
//   transB==1 : B is [N,K] row-major  (computes A * B^T)
// fp32 in/out, f16 WMMA compute, fp32 accumulate.
// Register-prefetch pipeline: global loads of tile k+1 overlap WMMA of tile k.
// Interior tiles take an unguarded fast path (batched loads, no branches).
// ---------------------------------------------------------------------------
__launch_bounds__(256)
__global__ void k_gemm(const float* __restrict__ A, int lda,
                       const float* __restrict__ B, int ldb, int transB,
                       float* __restrict__ Cp, int ldc,
                       int M, int N, int K,
                       const float* __restrict__ bias,
                       const float* __restrict__ resid,
                       float scale, int relu)
{
    __shared__ unsigned As[GT_M * LDSU];
    __shared__ unsigned Bs[GT_N * LDSU];

    const int tid  = threadIdx.x;
    const int lane = tid & 31;
    const int w    = tid >> 5;
    const int wm   = (w >> 1) * 32;   // 4 waves along M
    const int wn   = (w & 1) * 32;    // 2 waves along N
    const int bm   = blockIdx.y * GT_M;
    const int bn   = blockIdx.x * GT_N;

    v8f acc[2][2];
    v8f vz = {0.f,0.f,0.f,0.f,0.f,0.f,0.f,0.f};
    acc[0][0]=vz; acc[0][1]=vz; acc[1][0]=vz; acc[1][1]=vz;

    // ---- staging geometry (fixed per thread) ----
    const int arow = tid >> 1;
    const int akc  = (tid & 1) * 16;
    const int agr  = bm + arow;
    const int agrc = (agr < M) ? agr : (M - 1);
    const bool arok = (agr < M);
    const float* Arow = A + (size_t)agrc * lda;

    const bool bnFull = (bn + GT_N <= N);

    float ra[16];
    float rb[8];

    // ---- prefetch helpers ----
    auto loadA = [&](int k0) {
        if (arok && (k0 + GT_K <= K)) {
            const float* p = Arow + k0 + akc;
            #pragma unroll
            for (int j = 0; j < 16; ++j) ra[j] = p[j];
        } else {
            #pragma unroll
            for (int j = 0; j < 16; ++j) {
                int gk = k0 + akc + j;
                int c = (gk < K) ? gk : (K - 1);
                float v = Arow[c];
                ra[j] = (arok && gk < K) ? v : 0.f;
            }
        }
    };
    auto loadB = [&](int k0) {
        if (transB) {
            // B[N,K]: k-pair fastest across threads -> coalesced row reads
            if (bnFull && (k0 + GT_K <= K)) {
                #pragma unroll
                for (int i = 0; i < 4; ++i) {
                    int pidx = tid + i * 256;
                    int k2 = (pidx & 15) * 2;
                    int gn = bn + (pidx >> 4);
                    const float* p = B + (size_t)gn * ldb + k0 + k2;
                    rb[2*i]   = p[0];
                    rb[2*i+1] = p[1];
                }
            } else {
                #pragma unroll
                for (int i = 0; i < 4; ++i) {
                    int pidx = tid + i * 256;
                    int k2 = (pidx & 15) * 2;
                    int gn = bn + (pidx >> 4);
                    int gnc = (gn < N) ? gn : (N - 1);
                    const float* Br = B + (size_t)gnc * ldb;
                    int gk0 = k0 + k2, gk1 = gk0 + 1;
                    int c0 = (gk0 < K) ? gk0 : (K - 1);
                    int c1 = (gk1 < K) ? gk1 : (K - 1);
                    float v0 = Br[c0];
                    float v1 = Br[c1];
                    rb[2*i]   = (gn < N && gk0 < K) ? v0 : 0.f;
                    rb[2*i+1] = (gn < N && gk1 < K) ? v1 : 0.f;
                }
            }
        } else {
            // B[K,N]: n fastest across threads -> coalesced column reads
            if (bnFull && (k0 + GT_K <= K)) {
                #pragma unroll
                for (int i = 0; i < 4; ++i) {
                    int pidx = tid + i * 256;
                    int gn = bn + (pidx & 63);
                    int k2 = (pidx >> 6) * 2;
                    const float* p = B + (size_t)(k0 + k2) * ldb + gn;
                    rb[2*i]   = p[0];
                    rb[2*i+1] = p[ldb];
                }
            } else {
                #pragma unroll
                for (int i = 0; i < 4; ++i) {
                    int pidx = tid + i * 256;
                    int gn = bn + (pidx & 63);
                    int k2 = (pidx >> 6) * 2;
                    int cn = (gn < N) ? gn : (N - 1);
                    int gk0 = k0 + k2, gk1 = gk0 + 1;
                    int r0 = (gk0 < K) ? gk0 : (K - 1);
                    int r1 = (gk1 < K) ? gk1 : (K - 1);
                    float v0 = B[(size_t)r0 * ldb + cn];
                    float v1 = B[(size_t)r1 * ldb + cn];
                    rb[2*i]   = (gn < N && gk0 < K) ? v0 : 0.f;
                    rb[2*i+1] = (gn < N && gk1 < K) ? v1 : 0.f;
                }
            }
        }
    };

    // prime the pipeline
    loadA(0);
    loadB(0);

    for (int k0 = 0; k0 < K; k0 += GT_K) {
        // ---- commit prefetched registers into LDS (packed f16 dwords) ----
        #pragma unroll
        for (int j = 0; j < 8; ++j)
            As[arow * LDSU + (akc >> 1) + j] = packh(ra[2*j], ra[2*j+1]);
        if (transB) {
            #pragma unroll
            for (int i = 0; i < 4; ++i) {
                int pidx = tid + i * 256;
                int k2 = (pidx & 15) * 2;
                int nn = pidx >> 4;
                Bs[nn * LDSU + (k2 >> 1)] = packh(rb[2*i], rb[2*i+1]);
            }
        } else {
            #pragma unroll
            for (int i = 0; i < 4; ++i) {
                int pidx = tid + i * 256;
                int nn = pidx & 63;
                int k2 = (pidx >> 6) * 2;
                Bs[nn * LDSU + (k2 >> 1)] = packh(rb[2*i], rb[2*i+1]);
            }
        }
        __syncthreads();

        // ---- prefetch next tile (overlaps DS reads + WMMA below) ----
        int kn = k0 + GT_K;
        if (kn < K) {
            loadA(kn);
            loadB(kn);
        }

        // ---- build fragments (contiguous k = 16h + s, same for A and B) and issue 4 WMMAs ----
        const int r    = lane & 15;
        const int aoff = (lane & 16) ? 8 : 0;  // dword offset: K 16..31 for high lanes
        const int boff = (lane & 16) ? 8 : 0;  // dword offset: K 16..31 for high lanes
        v16h afrag[2], bfrag[2];
        #pragma unroll
        for (int t = 0; t < 2; ++t) {
            const unsigned* pa = As + (wm + t*16 + r) * LDSU + aoff;
            union { v16h v; unsigned u[8]; } UA;
            #pragma unroll
            for (int j = 0; j < 8; ++j) UA.u[j] = pa[j];
            afrag[t] = UA.v;
            const unsigned* pb = Bs + (wn + t*16 + r) * LDSU + boff;
            union { v16h v; unsigned u[8]; } UB;
            #pragma unroll
            for (int j = 0; j < 8; ++j) UB.u[j] = pb[j];
            bfrag[t] = UB.v;
        }
        #pragma unroll
        for (int i = 0; i < 2; ++i)
            #pragma unroll
            for (int j = 0; j < 2; ++j)
                acc[i][j] = __builtin_amdgcn_wmma_f32_16x16x32_f16(
                    false, afrag[i], false, bfrag[j], (short)0, acc[i][j], false, false);
        __syncthreads();
    }

    // ---- epilogue: scale, bias, residual, relu ----
    const int cn0 = bn + wn + (lane & 15);
    const int mb  = bm + wm + ((lane >> 4) << 3);
    #pragma unroll
    for (int j = 0; j < 2; ++j) {
        int cn = cn0 + j * 16;
        if (cn >= N) continue;
        float bv = bias ? bias[cn] : 0.f;
        #pragma unroll
        for (int i = 0; i < 2; ++i) {
            #pragma unroll
            for (int rr = 0; rr < 8; ++rr) {
                int cm = mb + i * 16 + rr;
                if (cm >= M) continue;
                float v = acc[i][j][rr] * scale + bv;
                if (resid) v += resid[(size_t)cm * ldc + cn];
                if (relu && v < 0.f) v = 0.f;
                Cp[(size_t)cm * ldc + cn] = v;
            }
        }
    }
}

// ---------------------------------------------------------------------------
// Elementwise / reduction / graph kernels
// ---------------------------------------------------------------------------
__global__ void k_zero(float* p, long n) {
    long i = (long)blockIdx.x * blockDim.x + threadIdx.x;
    if (i < n) p[i] = 0.f;
}
__global__ void k_fill_bias(float* out, const float* bias, const float* add,
                            int rows, int cols) {
    long i = (long)blockIdx.x * blockDim.x + threadIdx.x;
    if (i >= (long)rows * cols) return;
    int c = (int)(i % cols);
    float v = bias[c];
    if (add) v += add[i];
    out[i] = v;
}
__global__ void k_add2(float* out, const float* a, const float* b, long n) {
    long i = (long)blockIdx.x * blockDim.x + threadIdx.x;
    if (i < n) out[i] = a[i] + b[i];
}
__global__ void k_add3_scale(float* out, const float* a, const float* b,
                             const float* c, float s, long n) {
    long i = (long)blockIdx.x * blockDim.x + threadIdx.x;
    if (i < n) out[i] = (a[i] + b[i] + c[i]) * s;
}
__global__ void k_concat(float* t, const float* x, const float* se,
                         int N, int H, int sd) {
    int cw = H + sd;
    long i = (long)blockIdx.x * blockDim.x + threadIdx.x;
    if (i >= (long)N * cw) return;
    int n = (int)(i / cw), c = (int)(i % cw);
    t[i] = (c < H) ? x[(size_t)n * H + c] : se[(size_t)n * sd + (c - H)];
}
__global__ void k_copycols(float* dst, int dld, int coff, const float* src,
                           int rows, int cols) {
    long i = (long)blockIdx.x * blockDim.x + threadIdx.x;
    if (i >= (long)rows * cols) return;
    int r = (int)(i / cols), c = (int)(i % cols);
    dst[(size_t)r * dld + coff + c] = src[i];
}
__global__ void k_colstats(const float* h, int rows, int cols,
                           float* mu, float* rinv) {
    int c = blockIdx.x * blockDim.x + threadIdx.x;
    if (c >= cols) return;
    float s = 0.f, s2 = 0.f;
    for (int r = 0; r < rows; ++r) {
        float v = h[(size_t)r * cols + c];
        s += v; s2 += v * v;
    }
    float m = s / rows;
    float var = s2 / rows - m * m;
    mu[c] = m;
    rinv[c] = rsqrtf(var + 1e-5f);
}
__global__ void k_bn_apply(float* h, const float* mu, const float* rinv,
                           const float* g, const float* b, int rows, int cols) {
    long i = (long)blockIdx.x * blockDim.x + threadIdx.x;
    if (i >= (long)rows * cols) return;
    int c = (int)(i % cols);
    h[i] = g[c] * (h[i] - mu[c]) * rinv[c] + b[c];
}
__global__ void k_deg(const int* dstl, int E, int Nn, float* deg) {
    int e = blockIdx.x * blockDim.x + threadIdx.x;
    if (e >= E + Nn) return;
    int d = (e < E) ? dstl[e] : (e - E);
    atomicAdd(&deg[d], 1.f);
}
__global__ void k_dinv(const float* deg, float* drs, float* dinv, int Nn) {
    int n = blockIdx.x * blockDim.x + threadIdx.x;
    if (n >= Nn) return;
    float d = fmaxf(deg[n], 1.f);
    drs[n] = rsqrtf(d);
    dinv[n] = 1.f / d;
}
__global__ void k_gcn_scatter(const float* h, int ld, const int* srcl,
                              const int* dstl, int E, int Nn, const float* dr,
                              float* out, int cols) {
    int e = blockIdx.y;
    int c = blockIdx.x * blockDim.x + threadIdx.x;
    if (c >= cols) return;
    int s, d;
    if (e < E) { s = srcl[e]; d = dstl[e]; } else { s = d = e - E; }
    float coef = dr[s] * dr[d];
    atomicAdd(&out[(size_t)d * cols + c], h[(size_t)s * ld + c] * coef);
}
__global__ void k_scatter_sum(const float* h, int ld, const int* srcl,
                              const int* dstl, int E, int Nn,
                              float* out, int cols) {
    int e = blockIdx.y;
    int c = blockIdx.x * blockDim.x + threadIdx.x;
    if (c >= cols) return;
    int s, d;
    if (e < E) { s = srcl[e]; d = dstl[e]; } else { s = d = e - E; }
    atomicAdd(&out[(size_t)d * cols + c], h[(size_t)s * ld + c]);
}
__global__ void k_rowscale(float* h, const float* s, int rows, int cols) {
    long i = (long)blockIdx.x * blockDim.x + threadIdx.x;
    if (i >= (long)rows * cols) return;
    int r = (int)(i / cols);
    h[i] *= s[r];
}
__global__ void k_softmax_rows(float* att, int n, int ld) {
    __shared__ float red[256];
    float* p = att + (size_t)blockIdx.x * ld;
    int tid = threadIdx.x;
    float m = -1e30f;
    for (int c = tid; c < n; c += 256) m = fmaxf(m, p[c]);
    red[tid] = m; __syncthreads();
    for (int s = 128; s > 0; s >>= 1) {
        if (tid < s) red[tid] = fmaxf(red[tid], red[tid + s]);
        __syncthreads();
    }
    m = red[0]; __syncthreads();
    float sum = 0.f;
    for (int c = tid; c < n; c += 256) { float e = __expf(p[c] - m); p[c] = e; sum += e; }
    red[tid] = sum; __syncthreads();
    for (int s = 128; s > 0; s >>= 1) {
        if (tid < s) red[tid] += red[tid + s];
        __syncthreads();
    }
    float inv = 1.f / red[0];
    for (int c = tid; c < n; c += 256) p[c] *= inv;
}
// order-preserving float <-> u32 key for atomic segment-max
__device__ inline unsigned fkey(float x) {
    unsigned u = __float_as_uint(x);
    return (u & 0x80000000u) ? ~u : (u | 0x80000000u);
}
__device__ inline float funkey(unsigned k) {
    unsigned u = (k & 0x80000000u) ? (k ^ 0x80000000u) : ~k;
    return __uint_as_float(u);
}
__global__ void k_init_mkey(unsigned* mk, int Nn) {
    int n = blockIdx.x * blockDim.x + threadIdx.x;
    if (n < Nn) mk[n] = fkey(-1e30f);
}
__device__ inline float leaky02(float v) { return v > 0.f ? v : 0.2f * v; }
__global__ void k_gat_e1(const float* ps, const float* pd, const int* srcl,
                         const int* dstl, int E, int Nn, unsigned* mk) {
    int e = blockIdx.x * blockDim.x + threadIdx.x;
    if (e >= E + Nn) return;
    int s, d;
    if (e < E) { s = srcl[e]; d = dstl[e]; } else { s = d = e - E; }
    atomicMax(&mk[d], fkey(leaky02(ps[s] + pd[d])));
}
__global__ void k_gat_e2(const float* ps, const float* pd, const int* srcl,
                         const int* dstl, int E, int Nn, const unsigned* mk,
                         float* ex, float* ssum) {
    int e = blockIdx.x * blockDim.x + threadIdx.x;
    if (e >= E + Nn) return;
    int s, d;
    if (e < E) { s = srcl[e]; d = dstl[e]; } else { s = d = e - E; }
    float v = __expf(leaky02(ps[s] + pd[d]) - funkey(mk[d]));
    ex[e] = v;
    atomicAdd(&ssum[d], v);
}
__global__ void k_gat_scatter(const float* h, int ld, const int* srcl,
                              const int* dstl, int E, int Nn, const float* ex,
                              const float* ssum, float* out, int cols) {
    int e = blockIdx.y;
    int c = blockIdx.x * blockDim.x + threadIdx.x;
    if (c >= cols) return;
    int s, d;
    if (e < E) { s = srcl[e]; d = dstl[e]; } else { s = d = e - E; }
    float alpha = ex[e] / ssum[d];
    atomicAdd(&out[(size_t)d * cols + c], h[(size_t)s * ld + c] * alpha);
}
__global__ void k_pairs(const float* xc, const int* ei, const int* teid,
                        int EM, int cw, float* out) {
    int b = blockIdx.y;
    int c = blockIdx.x * blockDim.x + threadIdx.x;
    if (c >= cw) return;
    int id = teid[b];
    int n0 = ei[id], n1 = ei[EM + id];
    out[(size_t)b * cw + c] = xc[(size_t)n0 * cw + c] * xc[(size_t)n1 * cw + c];
}
// per-node dot products for GAT attention coefficients
__global__ void k_dots(const float* hg, const float* a1, const float* a2,
                       int Nn, int Hh, float* ps, float* pd) {
    int n = blockIdx.x * blockDim.x + threadIdx.x;
    if (n >= Nn) return;
    const float* r = hg + (size_t)n * Hh;
    float s1 = 0.f, s2 = 0.f;
    for (int c = 0; c < Hh; ++c) { float v = r[c]; s1 += v * a1[c]; s2 += v * a2[c]; }
    ps[n] = s1;
    pd[n] = s2;
}

// ---------------------------------------------------------------------------
// Host orchestration
// ---------------------------------------------------------------------------
extern "C" void kernel_launch(void* const* d_in, const int* in_sizes, int n_in,
                              void* d_out, int out_size, void* d_ws, size_t ws_size,
                              hipStream_t stream) {
    (void)in_sizes; (void)n_in; (void)out_size; (void)ws_size;

    const int Nn = 2048, Hh = 1024, SDd = 20, Cc = 1044, Kc = 7;
    const int Ee = 32768, EMm = 32768, Bp = 4096, DHh = 261;
    const int C3 = 3 * Cc, C2 = 2 * Cc, XCw = 8192;

    const float* X    = (const float*)d_in[0];
    const float* SE   = (const float*)d_in[1];
    const int*   SEI  = (const int*)  d_in[2];
    const int*   EI   = (const int*)  d_in[3];
    const int*   TEID = (const int*)  d_in[4];
    const float* gpsW_   = (const float*)d_in[5];
    const float* gpsB_   = (const float*)d_in[6];
    const float* aiW_    = (const float*)d_in[7];
    const float* aiB_    = (const float*)d_in[8];
    const float* aoW_    = (const float*)d_in[9];
    const float* aoB_    = (const float*)d_in[10];
    const float* bn1g_   = (const float*)d_in[11];
    const float* bn1b_   = (const float*)d_in[12];
    const float* bn2g_   = (const float*)d_in[13];
    const float* bn2b_   = (const float*)d_in[14];
    const float* bn3g_   = (const float*)d_in[15];
    const float* bn3b_   = (const float*)d_in[16];
    const float* mW1_    = (const float*)d_in[17];
    const float* mB1_    = (const float*)d_in[18];
    const float* mW2_    = (const float*)d_in[19];
    const float* mB2_    = (const float*)d_in[20];
    const float* seW_    = (const float*)d_in[21];
    const float* seB_    = (const float*)d_in[22];
    const float* gatW_   = (const float*)d_in[23];
    const float* gatAs_  = (const float*)d_in[24];
    const float* gatAd_  = (const float*)d_in[25];
    const float* gatB_   = (const float*)d_in[26];
    const float* subW_   = (const float*)d_in[27];
    const float* subB_   = (const float*)d_in[28];
    const float* gc2W_   = (const float*)d_in[29];
    const float* gc2B_   = (const float*)d_in[30];
    const float* bnfg_   = (const float*)d_in[31];
    const float* bnfb_   = (const float*)d_in[32];
    const float* clsW[6] = { (const float*)d_in[33], (const float*)d_in[35],
                             (const float*)d_in[37], (const float*)d_in[39],
                             (const float*)d_in[41], (const float*)d_in[43] };
    const float* clsB[6] = { (const float*)d_in[34], (const float*)d_in[36],
                             (const float*)d_in[38], (const float*)d_in[40],
                             (const float*)d_in[42], (const float*)d_in[44] };

    // workspace carve-up
    char* wsb = (char*)d_ws;
    size_t off = 0;
    auto alloc = [&](size_t nfl) -> float* {
        float* p = (float*)(wsb + off);
        off = (off + nfl * sizeof(float) + 255) & ~(size_t)255;
        return p;
    };
    float* t    = alloc((size_t)Nn * Cc);
    float* hA   = alloc((size_t)Nn * Cc);
    float* hB   = alloc((size_t)Nn * Cc);
    float* qkv  = alloc((size_t)Nn * C3);   // also reused as [N,C] GEMM temp
    float* att  = alloc((size_t)Nn * Nn);
    float* obuf = alloc((size_t)Nn * Cc);
    float* mlpb = alloc((size_t)Nn * C2);
    float* mu   = alloc(4096);
    float* rinv = alloc(4096);
    float* deg  = alloc(Nn);
    float* drs  = alloc(Nn);
    float* dinv = alloc(Nn);
    float* ps   = alloc(Nn);
    float* pd   = alloc(Nn);
    unsigned* mkey = (unsigned*)alloc(Nn);
    float* exbuf = alloc(Ee + Nn);
    float* ssum  = alloc(Nn);
    float* tH  = alloc((size_t)Nn * Hh);
    float* hg  = alloc((size_t)Nn * Hh);
    float* u1  = alloc((size_t)Nn * Hh);
    float* u2  = alloc((size_t)Nn * Hh);
    float* u3  = alloc((size_t)Nn * Hh);
    float* xc  = alloc((size_t)Nn * XCw);
    float* pa  = alloc((size_t)Bp * XCw);
    float* pb  = alloc((size_t)Bp * 4096);

    auto g1 = [](long n) { return dim3((unsigned)((n + 255) / 256)); };
    auto gemm = [&](const float* A, int lda, const float* B, int ldb, int tB,
                    float* C, int ldc, int M, int N, int K, const float* bias,
                    const float* resid, float scale, int relu) {
        dim3 g((N + GT_N - 1) / GT_N, (M + GT_M - 1) / GT_M);
        k_gemm<<<g, 256, 0, stream>>>(A, lda, B, ldb, tB, C, ldc, M, N, K,
                                      bias, resid, scale, relu);
    };
    auto bn = [&](float* h, int rows, int cols, const float* g, const float* b) {
        k_colstats<<<g1(cols), 256, 0, stream>>>(h, rows, cols, mu, rinv);
        k_bn_apply<<<g1((long)rows * cols), 256, 0, stream>>>(h, mu, rinv, g, b,
                                                              rows, cols);
    };

    // xc[:, 0:1024] = x
    k_copycols<<<g1((long)Nn * Hh), 256, 0, stream>>>(xc, XCw, 0, X, Nn, Hh);

    const float iscale = 1.0f / sqrtf((float)DHh);

    for (int i = 0; i < Kc; ++i) {
        const int* srcl = SEI + (size_t)i * 2 * Ee;
        const int* dstl = srcl + Ee;
        const float* gpsW = gpsW_ + (size_t)i * Cc * Cc;
        const float* gpsB = gpsB_ + (size_t)i * Cc;
        const float* aiW  = aiW_  + (size_t)i * C3 * Cc;
        const float* aiB  = aiB_  + (size_t)i * C3;
        const float* aoW  = aoW_  + (size_t)i * Cc * Cc;
        const float* aoB  = aoB_  + (size_t)i * Cc;
        const float* mW1  = mW1_  + (size_t)i * Cc * C2;
        const float* mB1  = mB1_  + (size_t)i * C2;
        const float* mW2  = mW2_  + (size_t)i * C2 * Cc;
        const float* mB2  = mB2_  + (size_t)i * Cc;
        const float* seW  = seW_  + (size_t)i * Cc * Hh;
        const float* seB  = seB_  + (size_t)i * Hh;
        const float* gatW = gatW_ + (size_t)i * Hh * Hh;
        const float* subW = subW_ + (size_t)i * Hh * Hh;
        const float* gc2W = gc2W_ + (size_t)i * Hh * Hh;

        // t = concat(x, se[i])
        k_concat<<<g1((long)Nn * Cc), 256, 0, stream>>>(
            t, X, SE + (size_t)i * Nn * SDd, Nn, Hh, SDd);

        // ---- MHA branch ----
        gemm(t, Cc, aiW, Cc, 1, qkv, C3, Nn, C3, Cc, aiB, nullptr, 1.f, 0);
        for (int h = 0; h < 4; ++h) {
            gemm(qkv + (size_t)h * DHh, C3, qkv + Cc + (size_t)h * DHh, C3, 1,
                 att, Nn, Nn, Nn, DHh, nullptr, nullptr, iscale, 0);
            k_softmax_rows<<<Nn, 256, 0, stream>>>(att, Nn, Nn);
            gemm(att, Nn, qkv + 2 * Cc + (size_t)h * DHh, C3, 0,
                 obuf + (size_t)h * DHh, Cc, Nn, DHh, Nn, nullptr, nullptr, 1.f, 0);
        }
        gemm(obuf, Cc, aoW, Cc, 1, hB, Cc, Nn, Cc, Cc, aoB, t, 1.f, 0);
        bn(hB, Nn, Cc, bn2g_ + (size_t)i * Cc, bn2b_ + (size_t)i * Cc);

        // ---- GCN branch (qkv reused as [N,C] temp) ----
        gemm(t, Cc, gpsW, Cc, 0, qkv, Cc, Nn, Cc, Cc, nullptr, nullptr, 1.f, 0);
        k_zero<<<g1(Nn), 256, 0, stream>>>(deg, Nn);
        k_deg<<<g1(Ee + Nn), 256, 0, stream>>>(dstl, Ee, Nn, deg);
        k_dinv<<<g1(Nn), 256, 0, stream>>>(deg, drs, dinv, Nn);
        k_fill_bias<<<g1((long)Nn * Cc), 256, 0, stream>>>(hA, gpsB, t, Nn, Cc);
        {
            dim3 gs((Cc + 255) / 256, Ee + Nn);
            k_gcn_scatter<<<gs, 256, 0, stream>>>(qkv, Cc, srcl, dstl, Ee, Nn,
                                                  drs, hA, Cc);
        }
        bn(hA, Nn, Cc, bn1g_ + (size_t)i * Cc, bn1b_ + (size_t)i * Cc);

        // h = h_local + h_attn; MLP with residual; bn3
        k_add2<<<g1((long)Nn * Cc), 256, 0, stream>>>(hA, hA, hB, (long)Nn * Cc);
        gemm(hA, Cc, mW1, C2, 0, mlpb, C2, Nn, C2, Cc, mB1, nullptr, 1.f, 1);
        gemm(mlpb, C2, mW2, Cc, 0, hB, Cc, Nn, Cc, C2, mB2, hA, 1.f, 0);
        bn(hB, Nn, Cc, bn3g_ + (size_t)i * Cc, bn3b_ + (size_t)i * Cc);

        // t = h @ se_lin + b
        gemm(hB, Cc, seW, Hh, 0, tH, Hh, Nn, Hh, Cc, seB, nullptr, 1.f, 0);

        // ---- GAT ----
        gemm(tH, Hh, gatW, Hh, 0, hg, Hh, Nn, Hh, Hh, nullptr, nullptr, 1.f, 0);
        k_dots<<<g1(Nn), 256, 0, stream>>>(hg, gatAs_ + (size_t)i * Hh,
                                           gatAd_ + (size_t)i * Hh, Nn, Hh, ps, pd);
        k_init_mkey<<<g1(Nn), 256, 0, stream>>>(mkey, Nn);
        k_gat_e1<<<g1(Ee + Nn), 256, 0, stream>>>(ps, pd, srcl, dstl, Ee, Nn, mkey);
        k_zero<<<g1(Nn), 256, 0, stream>>>(ssum, Nn);
        k_gat_e2<<<g1(Ee + Nn), 256, 0, stream>>>(ps, pd, srcl, dstl, Ee, Nn,
                                                  mkey, exbuf, ssum);
        k_fill_bias<<<g1((long)Nn * Hh), 256, 0, stream>>>(
            tH, gatB_ + (size_t)i * Hh, nullptr, Nn, Hh);
        {
            dim3 gs((Hh + 255) / 256, Ee + Nn);
            k_gat_scatter<<<gs, 256, 0, stream>>>(hg, Hh, srcl, dstl, Ee, Nn,
                                                  exbuf, ssum, tH, Hh);
        }

        // ---- two-hop mean aggregation ----
        {
            dim3 gs((Hh + 255) / 256, Ee + Nn);
            k_zero<<<g1((long)Nn * Hh), 256, 0, stream>>>(u1, (long)Nn * Hh);
            k_scatter_sum<<<gs, 256, 0, stream>>>(tH, Hh, srcl, dstl, Ee, Nn, u1, Hh);
            k_rowscale<<<g1((long)Nn * Hh), 256, 0, stream>>>(u1, dinv, Nn, Hh);
            k_zero<<<g1((long)Nn * Hh), 256, 0, stream>>>(u2, (long)Nn * Hh);
            k_scatter_sum<<<gs, 256, 0, stream>>>(u1, Hh, srcl, dstl, Ee, Nn, u2, Hh);
            k_rowscale<<<g1((long)Nn * Hh), 256, 0, stream>>>(u2, dinv, Nn, Hh);
        }
        k_add3_scale<<<g1((long)Nn * Hh), 256, 0, stream>>>(
            u3, tH, u1, u2, 1.f / 3.f, (long)Nn * Hh);
        gemm(u3, Hh, subW, Hh, 0, u1, Hh, Nn, Hh, Hh,
             subB_ + (size_t)i * Hh, nullptr, 1.f, 1);  // t_sub in u1

        // ---- final GCN + t_sub, bnf ----
        gemm(tH, Hh, gc2W, Hh, 0, hg, Hh, Nn, Hh, Hh, nullptr, nullptr, 1.f, 0);
        k_fill_bias<<<g1((long)Nn * Hh), 256, 0, stream>>>(
            u2, gc2B_ + (size_t)i * Hh, u1, Nn, Hh);
        {
            dim3 gs((Hh + 255) / 256, Ee + Nn);
            k_gcn_scatter<<<gs, 256, 0, stream>>>(hg, Hh, srcl, dstl, Ee, Nn,
                                                  drs, u2, Hh);
        }
        bn(u2, Nn, Hh, bnfg_ + (size_t)i * Hh, bnfb_ + (size_t)i * Hh);
        k_copycols<<<g1((long)Nn * Hh), 256, 0, stream>>>(
            xc, XCw, Hh * (i + 1), u2, Nn, Hh);
    }

    // pair features
    {
        dim3 gs((XCw + 255) / 256, Bp);
        k_pairs<<<gs, 256, 0, stream>>>(xc, EI, TEID, EMm, XCw, pa);
    }
    // classifier chain (pure affine)
    gemm(pa, 8192, clsW[0], 4096, 0, pb, 4096, Bp, 4096, 8192, clsB[0], nullptr, 1.f, 0);
    gemm(pb, 4096, clsW[1], 2048, 0, pa, 2048, Bp, 2048, 4096, clsB[1], nullptr, 1.f, 0);
    gemm(pa, 2048, clsW[2], 1024, 0, pb, 1024, Bp, 1024, 2048, clsB[2], nullptr, 1.f, 0);
    gemm(pb, 1024, clsW[3],  512, 0, pa,  512, Bp,  512, 1024, clsB[3], nullptr, 1.f, 0);
    gemm(pa,  512, clsW[4],  256, 0, pb,  256, Bp,  256,  512, clsB[4], nullptr, 1.f, 0);
    gemm(pb,  256, clsW[5],    7, 0, (float*)d_out, 7, Bp, 7, 256, clsB[5], nullptr, 1.f, 0);
}
